// RNN_50680614093089
// MI455X (gfx1250) — matrix operand order repaired and checked by reading, not compiled, not run on hardware
//
#include <hip/hip_runtime.h>
#include <hip/hip_bf16.h>
#include <stdint.h>

#define N_HIDDEN 1024
#define N_INPUTS 8
#define N_OUTPUTS 8
#define BATCH 32
#define SEQ_LEN 1000

__device__ __constant__ float c_DT   = 0.001f;
__device__ __constant__ float c_ITAU = 100.0f;    // 1/TAU
__device__ __constant__ float c_ITD  = 5.0f;      // 1/TAU_DEPR
__device__ __constant__ float c_ITF  = 0.6666666666666666f; // 1/TAU_FACIL
__device__ __constant__ float c_BETA = 10.0f;

typedef __attribute__((ext_vector_type(16))) _Float16 v16h;
typedef __attribute__((ext_vector_type(8)))  float    v8f;
typedef __attribute__((ext_vector_type(4)))  unsigned uv4;
typedef __attribute__((ext_vector_type(4)))  float    fv4;

union Frag { struct { uv4 lo, hi; } q; uint32_t u[8]; v16h v; };

__device__ inline uint32_t pack2h(float a, float b) {
  union { _Float16 h[2]; uint32_t u; } p;
  p.h[0] = (_Float16)a; p.h[1] = (_Float16)b; return p.u;
}

// ---- workspace layout (bytes) ----
// [0, 16384)            : barrier counters (4096 u32)
// [16384, 147456)       : double-buffered activation a[2][32][1024] f16
// [147456, 2244608)     : W_eff packed B-fragments, 64 slices * 8192 dwords
// [2244608, 2277376)    : W_hz packed B-fragments, 8192 dwords

// --------------------------------------------------------------------------
// Kernel 1: zero barrier counters; pack W_eff = W_hh*mask and W_hz into
// WMMA B-operand fragment layout (f16). B frag (32x16): lane=N (l&15), dword v = K pair (table order).
// --------------------------------------------------------------------------
__global__ void prep_kernel(const float* __restrict__ Whh,
                            const float* __restrict__ Wmask,
                            const float* __restrict__ Whz,
                            unsigned* __restrict__ cnt,
                            uint32_t* __restrict__ wb,
                            uint32_t* __restrict__ wzb) {
  int i0 = blockIdx.x * blockDim.x + threadIdx.x;
  int stride = gridDim.x * blockDim.x;
  for (int i = i0; i < 4096; i += stride) cnt[i] = 0u;
  for (int i = i0; i < 64 * 32 * 32 * 8; i += stride) {
    int v = i & 7, l = (i >> 3) & 31, kk = (i >> 8) & 31, g = i >> 13;
    int k = kk * 32 + ((v & 4) << 2) + (l >> 4) * 8 + (v & 3) * 2;
    int n = g * 16 + (l & 15);
    float a = Whh[(size_t)n * 1024 + k] * Wmask[(size_t)n * 1024 + k];
    float b = Whh[(size_t)n * 1024 + k + 1] * Wmask[(size_t)n * 1024 + k + 1];
    wb[i] = pack2h(a, b);
  }
  for (int i = i0; i < 32 * 32 * 8; i += stride) {
    int v = i & 7, l = (i >> 3) & 31, kk = i >> 8;
    int k = kk * 32 + ((v & 4) << 2) + (l >> 4) * 8 + (v & 3) * 2;
    int n = l & 15;
    float a = (n < N_OUTPUTS) ? Whz[(size_t)n * 1024 + k]     : 0.0f;
    float b = (n < N_OUTPUTS) ? Whz[(size_t)n * 1024 + k + 1] : 0.0f;
    wzb[i] = pack2h(a, b);
  }
}

// --------------------------------------------------------------------------
// Grid-wide barrier: one fresh counter slot per step (no reset hazard).
// --------------------------------------------------------------------------
__device__ inline void grid_barrier(unsigned* cnt, int slot, unsigned nblk) {
  __syncthreads();
  if (threadIdx.x == 0) {
    __hip_atomic_fetch_add(&cnt[slot], 1u, __ATOMIC_ACQ_REL, __HIP_MEMORY_SCOPE_AGENT);
    while (__hip_atomic_load(&cnt[slot], __ATOMIC_ACQUIRE, __HIP_MEMORY_SCOPE_AGENT) < nblk)
      __builtin_amdgcn_s_sleep(1);
  }
  __syncthreads();
}

// --------------------------------------------------------------------------
// Kernel 2: persistent scan. 64 WGs x 64 threads (2 waves). WG g owns hidden
// columns [16g, 16g+16). Wave w owns batches [16w, 16w+16). W slice lives in
// LDS (packed B frags); h/r/u state lives in registers matching the WMMA C
// layout (lane l, acc v -> batch 16w + v + 8*(l>>4), col 16g + (l&15)).
// Activation vector a passes between steps via a double-buffered f16 global
// buffer (L2-resident); a grid barrier separates steps.
// --------------------------------------------------------------------------
__global__ void __launch_bounds__(64) scan_kernel(
    const float* __restrict__ x,  const float* __restrict__ h0,
    const float* __restrict__ r0, const float* __restrict__ u0,
    const float* __restrict__ p_rel, const float* __restrict__ scal,
    const float* __restrict__ Wih, const uint32_t* __restrict__ wb,
    unsigned* cnt, _Float16* abuf, float* __restrict__ out) {
  __shared__ uv4 sW[2048];                       // 32 KB W slice (B frags)
  const int g = blockIdx.x;
  const int tid = threadIdx.x;
  const int w = tid >> 5, l = tid & 31;
  const int half = l >> 4, li = l & 15;
  const int j = g * 16 + li;                      // owned hidden column

  // stage W slice: global -> LDS once, reused 1000x by both waves
  const uv4* wb4 = (const uv4*)(wb + (size_t)g * 8192);
  for (int i = tid; i < 2048; i += 64) sW[i] = wb4[i];

  const float p = p_rel[j], sc = scal[j];
  float wih[8];
#pragma unroll
  for (int i = 0; i < 8; i++) wih[i] = Wih[(size_t)j * 8 + i];

  float hS[8], rS[8], uS[8], htr[8];
  int bidx[8];
#pragma unroll
  for (int v = 0; v < 8; v++) {
    int b = w * 16 + v + 8 * half;                // matches C-matrix layout
    bidx[v] = b;
    hS[v] = h0[(size_t)b * 1024 + j];
    rS[v] = r0[(size_t)b * 1024 + j];
    uS[v] = u0[(size_t)b * 1024 + j];
    htr[v] = tanhf(hS[v]);
    abuf[(size_t)b * 1024 + j] = (_Float16)(rS[v] * sc * htr[v]); // a_0
  }

  float* outH = out + (size_t)BATCH * SEQ_LEN * N_OUTPUTS;
  float* outR = outH + (size_t)BATCH * SEQ_LEN * N_HIDDEN;
  float* outU = outR + (size_t)BATCH * SEQ_LEN * N_HIDDEN;

  grid_barrier(cnt, 0, 64u);                      // a_0 visible everywhere
  int cur = 0;
  for (int t = 0; t < SEQ_LEN; t++) {
    const _Float16* A = abuf + (size_t)cur * (BATCH * N_HIDDEN);
    _Float16* An      = abuf + (size_t)(cur ^ 1) * (BATCH * N_HIDDEN);

    v8f acc = {};
    for (int kk = 0; kk < 32; kk++) {
      // A frag (16-bit 16x32): lane holds row m = 16w + li; two 16B K-chunks
      Frag fa, fb;
      const uv4* ap = (const uv4*)(A + (size_t)(w * 16 + li) * 1024
                                     + kk * 32 + half * 8);
      fa.q.lo = ap[0];                            // K = kb + half*8 .. +7
      fa.q.hi = ap[2];                            // K = kb + 16 + half*8 .. +7
      fb.q.lo = sW[(kk * 32 + l) * 2];            // B frag: lane = N
      fb.q.hi = sW[(kk * 32 + l) * 2 + 1];
      acc = __builtin_amdgcn_wmma_f32_16x16x32_f16(
          false, fa.v, false, fb.v, (short)0, acc, false, false);
    }

#pragma unroll
    for (int v = 0; v < 8; v++) {
      int b = bidx[v];
      const fv4* xr = (const fv4*)(x + ((size_t)b * SEQ_LEN + t) * 8);
      fv4 x0 = xr[0], x1 = xr[1];
      float xp = x0.x * wih[0] + x0.y * wih[1] + x0.z * wih[2] + x0.w * wih[3]
               + x1.x * wih[4] + x1.y * wih[5] + x1.z * wih[6] + x1.w * wih[7];
      float hn = hS[v] + c_DT * c_ITAU * (-hS[v] + xp + acc[v]);
      float drive = 0.5f * (1.0f + htr[v]);
      float rn = rS[v] + ((p - rS[v]) * c_ITD - c_BETA * rS[v] * drive) * c_DT;
      float un = uS[v] + ((p - uS[v]) * c_ITF + c_BETA * (1.0f - uS[v]) * drive) * c_DT;
      float ht = tanhf(hn);
      size_t o = ((size_t)b * SEQ_LEN + t) * 1024 + j;
      outH[o] = hn; outR[o] = rn; outU[o] = un;
      An[(size_t)b * 1024 + j] = (_Float16)(rn * sc * ht);
      hS[v] = hn; rS[v] = rn; uS[v] = un; htr[v] = ht;
    }
    if (t + 1 < SEQ_LEN)
      __builtin_prefetch(x + ((size_t)bidx[0] * SEQ_LEN + (t + 1)) * 8, 0, 1);
    grid_barrier(cnt, t + 1, 64u);
    cur ^= 1;
  }
}

// --------------------------------------------------------------------------
// Kernel 3: z = tanh(h) @ W_hz.T, hoisted out of the scan (parallel over all
// 32000 (b,t) rows). WMMA with M = (b,t) row tile, N = 16 (8 real outputs).
// --------------------------------------------------------------------------
__global__ void __launch_bounds__(128) zout_kernel(const uint32_t* __restrict__ wzb,
                                                   float* __restrict__ out) {
  __shared__ uv4 sW[2048];
  const int tid = threadIdx.x;
  const int w = tid >> 5, l = tid & 31;
  const int half = l >> 4, li = l & 15;
  const uv4* w4 = (const uv4*)wzb;
  for (int i = tid; i < 2048; i += 128) sW[i] = w4[i];
  __syncthreads();

  const float* H = out + (size_t)BATCH * SEQ_LEN * N_OUTPUTS;
  const int row0 = (blockIdx.x * 4 + w) * 16;

  v8f acc = {};
  for (int kk = 0; kk < 32; kk++) {
    const float* hp = H + (size_t)(row0 + li) * 1024 + kk * 32 + half * 8;
    fv4 c0 = ((const fv4*)hp)[0], c1 = ((const fv4*)hp)[1];
    fv4 c2 = ((const fv4*)(hp + 16))[0], c3 = ((const fv4*)(hp + 16))[1];
    Frag fa, fb;
    fa.u[0] = pack2h(tanhf(c0.x), tanhf(c0.y));
    fa.u[1] = pack2h(tanhf(c0.z), tanhf(c0.w));
    fa.u[2] = pack2h(tanhf(c1.x), tanhf(c1.y));
    fa.u[3] = pack2h(tanhf(c1.z), tanhf(c1.w));
    fa.u[4] = pack2h(tanhf(c2.x), tanhf(c2.y));
    fa.u[5] = pack2h(tanhf(c2.z), tanhf(c2.w));
    fa.u[6] = pack2h(tanhf(c3.x), tanhf(c3.y));
    fa.u[7] = pack2h(tanhf(c3.z), tanhf(c3.w));
    fb.q.lo = sW[(kk * 32 + l) * 2];
    fb.q.hi = sW[(kk * 32 + l) * 2 + 1];
    acc = __builtin_amdgcn_wmma_f32_16x16x32_f16(
        false, fa.v, false, fb.v, (short)0, acc, false, false);
  }
  if (li < N_OUTPUTS) {
#pragma unroll
    for (int v = 0; v < 8; v++) {
      int row = row0 + v + 8 * half;
      out[(size_t)row * N_OUTPUTS + li] = acc[v];
    }
  }
}

extern "C" void kernel_launch(void* const* d_in, const int* in_sizes, int n_in,
                              void* d_out, int out_size, void* d_ws, size_t ws_size,
                              hipStream_t stream) {
  (void)in_sizes; (void)n_in; (void)out_size; (void)ws_size;
  const float* x     = (const float*)d_in[0];
  const float* h0    = (const float*)d_in[1];
  const float* r0    = (const float*)d_in[2];
  const float* u0    = (const float*)d_in[3];
  const float* p_rel = (const float*)d_in[4];
  const float* scal  = (const float*)d_in[5];
  const float* Wih   = (const float*)d_in[6];
  const float* Whh   = (const float*)d_in[7];
  const float* Wmask = (const float*)d_in[8];
  const float* Whz   = (const float*)d_in[9];

  char* ws = (char*)d_ws;
  unsigned* cnt  = (unsigned*)(ws + 0);
  _Float16* abuf = (_Float16*)(ws + 16384);
  uint32_t* wb   = (uint32_t*)(ws + 147456);
  uint32_t* wzb  = (uint32_t*)(ws + 2244608);
  float* out = (float*)d_out;

  prep_kernel<<<512, 256, 0, stream>>>(Whh, Wmask, Whz, cnt, wb, wzb);
  scan_kernel<<<64, 64, 0, stream>>>(x, h0, r0, u0, p_rel, scal, Wih, wb, cnt, abuf, out);
  zout_kernel<<<500, 128, 0, stream>>>(wzb, out);
}
